// SigmoidMoE_592705486934
// MI455X (gfx1250) — matrix operand order repaired and checked by reading, not compiled, not run on hardware
//
#include <hip/hip_runtime.h>

// ---------------------------------------------------------------------------
// Sigmoid-MoE (top-2 of 8 experts), SwiGLU experts, bf16 WMMA on gfx1250.
// B=2,S=1024 -> 2048 tokens; D=1024; H=4096; E=8; NUM_ACTIVE=2.
// GEMM tiles: 128x64, K-step 32, 4 waves, 32 rows/wave (2 A-frags).
// A tiles staged with global_load_async_to_lds_b128 (ASYNCcnt).
// ---------------------------------------------------------------------------

typedef float        f32x4  __attribute__((ext_vector_type(4)));
typedef float        f32x8  __attribute__((ext_vector_type(8)));
typedef unsigned int u32x2  __attribute__((ext_vector_type(2)));
typedef unsigned int u32x4  __attribute__((ext_vector_type(4)));
typedef __bf16       bf16x16 __attribute__((ext_vector_type(16)));

union Frag { u32x4 u[2]; bf16x16 v; };

constexpr int TN = 2048;          // tokens (B*S)
constexpr int DD = 1024;          // model dim
constexpr int HH = 4096;          // hidden dim
constexpr int EE = 8;             // experts
constexpr int SLOTS = TN * 2;     // total routed assignments (top-2)

constexpr int MT = 128, NT = 64, KT = 32;  // GEMM tile
constexpr int AS = 40;                     // LDS row stride (shorts) = 80B, conflict-free

// workspace layout (bytes)
constexpr size_t XB_OFF   = 0;                                   // x as bf16
constexpr size_t XB_BYTES = (size_t)TN * DD * 2;                 // 4 MB
constexpr size_t HB_OFF   = XB_OFF + XB_BYTES;                   // h as bf16
constexpr size_t HB_BYTES = (size_t)SLOTS * HH * 2;              // 32 MB
constexpr size_t GT_OFF   = HB_OFF + HB_BYTES;                   // gates (f32)
constexpr size_t GT_BYTES = (size_t)SLOTS * 4;
constexpr size_t TK_OFF   = GT_OFF + GT_BYTES;                   // token ids
constexpr size_t TK_BYTES = (size_t)SLOTS * 4;
constexpr size_t T2W_OFF  = TK_OFF + TK_BYTES;                   // top-2 weights
constexpr size_t T2W_BYTES= (size_t)TN * 2 * 4;
constexpr size_t T2I_OFF  = T2W_OFF + T2W_BYTES;                 // top-2 expert ids
constexpr size_t T2I_BYTES= (size_t)TN * 2 * 4;
constexpr size_t META_OFF = T2I_OFF + T2I_BYTES;                 // counts[8],offs[8],cursor[8]
constexpr size_t AUX_OFF  = META_OFF + 32 * sizeof(int);         // aux-loss accumulator

__device__ __forceinline__ unsigned short f2bf(float f) {
  unsigned int u = __float_as_uint(f);
  unsigned int r = ((u >> 16) & 1u) + 0x7FFFu;   // round-to-nearest-even
  return (unsigned short)((u + r) >> 16);
}

__device__ __forceinline__ unsigned pk2bf(float lo, float hi) {
  return (unsigned)f2bf(lo) | ((unsigned)f2bf(hi) << 16);
}

// async DMA: 16 bytes global -> LDS, tracked by ASYNCcnt
__device__ __forceinline__ void async_copy16(const void* g, void* l) {
  unsigned lofs = (unsigned)(size_t)l;                 // LDS aperture: low 32b = offset
  unsigned long long ga = (unsigned long long)(size_t)g;
  asm volatile("global_load_async_to_lds_b128 %0, %1, off"
               :: "v"(lofs), "v"(ga) : "memory");
}
__device__ __forceinline__ void async_wait0() {
  asm volatile("s_wait_asynccnt 0x0" ::: "memory");
}

// ---------------------------------------------------------------------------
__global__ void moe_init(float* __restrict__ out, int n,
                         int* __restrict__ meta, float* __restrict__ aux) {
  const int i = blockIdx.x * blockDim.x + threadIdx.x;
  if (i < n) out[i] = 0.f;
  if (i < EE) meta[i] = 0;
  if (i == 0) *aux = 0.f;
}

__global__ void moe_xcvt(const float* __restrict__ x,
                         unsigned short* __restrict__ xb, int n4) {
  const int g = blockIdx.x * blockDim.x + threadIdx.x;
  if (g >= n4) return;
  f32x4 v = *(const f32x4*)(x + (size_t)g * 4);
  u32x2 o;
  o[0] = pk2bf(v[0], v[1]);
  o[1] = pk2bf(v[2], v[3]);
  *(u32x2*)(xb + (size_t)g * 4) = o;
}

// one wave32 per token: logits, sigmoid, top-2, expert counts, aux loss
__global__ __launch_bounds__(256) void moe_router(
    const float* __restrict__ x, const float* __restrict__ rw,
    const float* __restrict__ rb, float* __restrict__ t2w,
    int* __restrict__ t2i, int* __restrict__ meta, float* __restrict__ aux) {
  const int lane  = threadIdx.x & 31;
  const int token = blockIdx.x * 8 + (threadIdx.x >> 5);
  if (token >= TN) return;
  const float* xrow = x + (size_t)token * DD;
  float acc[EE];
  #pragma unroll
  for (int e = 0; e < EE; ++e) acc[e] = 0.f;
  for (int i = 0; i < 8; ++i) {
    const int d = lane * 4 + i * 128;
    f32x4 xv = *(const f32x4*)(xrow + d);
    #pragma unroll
    for (int j = 0; j < 4; ++j) {
      const f32x4* wr = (const f32x4*)(rw + (size_t)(d + j) * EE);
      f32x4 w0 = wr[0], w1 = wr[1];
      #pragma unroll
      for (int q = 0; q < 4; ++q) {
        acc[q]     += xv[j] * w0[q];
        acc[4 + q] += xv[j] * w1[q];
      }
    }
  }
  #pragma unroll
  for (int e = 0; e < EE; ++e)
    for (int o = 16; o > 0; o >>= 1) acc[e] += __shfl_down(acc[e], o, 32);
  if (lane == 0) {
    float s[EE], ss = 0.f;
    #pragma unroll
    for (int e = 0; e < EE; ++e) {
      float lg = acc[e] + rb[e];
      ss += lg * lg;
      s[e] = 1.f / (1.f + __expf(-lg));
    }
    int i0 = 0;
    #pragma unroll
    for (int e = 1; e < EE; ++e) if (s[e] > s[i0]) i0 = e;
    int i1 = (i0 == 0) ? 1 : 0;
    #pragma unroll
    for (int e = 0; e < EE; ++e) if (e != i0 && s[e] > s[i1]) i1 = e;
    const float rs = 1.f / (s[i0] + s[i1] + 1e-6f);
    t2i[2 * token] = i0;  t2i[2 * token + 1] = i1;
    t2w[2 * token] = s[i0] * rs;  t2w[2 * token + 1] = s[i1] * rs;
    atomicAdd(&meta[i0], 1);
    atomicAdd(&meta[i1], 1);
    atomicAdd(aux, ss);
  }
}

__global__ void moe_prefix(int* meta) {
  if (threadIdx.x == 0 && blockIdx.x == 0) {
    int o = 0;
    for (int e = 0; e < EE; ++e) { meta[8 + e] = o; meta[16 + e] = o; o += meta[e]; }
  }
}

__global__ __launch_bounds__(256) void moe_scatter(
    const int* __restrict__ t2i, const float* __restrict__ t2w,
    int* __restrict__ meta, int* __restrict__ toks, float* __restrict__ gates) {
  const int t = blockIdx.x * blockDim.x + threadIdx.x;
  if (t >= TN) return;
  #pragma unroll
  for (int k = 0; k < 2; ++k) {
    const int e = t2i[2 * t + k];
    const int slot = atomicAdd(&meta[16 + e], 1);
    toks[slot] = t;
    gates[slot] = t2w[2 * t + k];
  }
}

// ---------------------------------------------------------------------------
// GEMM1: gathered x[cnt,1024] @ w12[e][1024,8192] -> h = silu(x1)*x2 (bf16)
// ---------------------------------------------------------------------------
__global__ __launch_bounds__(128) void moe_gemm1(
    const unsigned short* __restrict__ xb, const float* __restrict__ w12,
    const int* __restrict__ toks, const int* __restrict__ meta,
    unsigned short* __restrict__ hbuf) {
  const int e = blockIdx.z;
  const int cnt = meta[e];
  const int rowBase = blockIdx.y * MT;
  if (rowBase >= cnt) return;             // sparse early-exit
  const int off = meta[8 + e];
  const int n0 = blockIdx.x * NT;

  __shared__ __align__(16) unsigned short lA [MT * AS];
  __shared__ __align__(16) unsigned short lB1[NT * AS];
  __shared__ __align__(16) unsigned short lB2[NT * AS];

  const int tid = threadIdx.x;
  const int w = tid >> 5, lane = tid & 31;

  f32x8 acc1[2][4], acc2[2][4];
  #pragma unroll
  for (int m = 0; m < 2; ++m)
    #pragma unroll
    for (int i = 0; i < 4; ++i)
      #pragma unroll
      for (int j = 0; j < 8; ++j) { acc1[m][i][j] = 0.f; acc2[m][i][j] = 0.f; }

  // A-tile fill: one thread per gathered row (async DMA, bf16 already)
  int gr = rowBase + tid; if (gr >= cnt) gr = cnt - 1;   // clamp padded rows
  const unsigned short* xrow = xb + (size_t)toks[off + gr] * DD;

  // B-tile fill: thread handles a k-pair x 4n quad
  const int bp = tid >> 4;          // k-pair index 0..7 (x2 passes)
  const int bn = (tid & 15) * 4;    // n quad
  const float* wbase = w12 + (size_t)e * DD * (2 * HH);

  for (int kt = 0; kt < DD / KT; ++kt) {
    const int k0 = kt * KT;
    __syncthreads();
    #pragma unroll
    for (int c = 0; c < 4; ++c)     // A: 64B/row via async DMA
      async_copy16(xrow + k0 + c * 8, lA + tid * AS + c * 8);
    #pragma unroll
    for (int r = 0; r < 2; ++r) {   // B: fp32 -> packed bf16, transposed
      const int k = 2 * (bp + r * 8);
      const float* p0 = wbase + (size_t)(k0 + k) * (2 * HH) + (n0 + bn);
      const float* p1 = p0 + 2 * HH;
      f32x4 a1 = *(const f32x4*)p0,        b1 = *(const f32x4*)p1;        // w1 rows k,k+1
      f32x4 a2 = *(const f32x4*)(p0 + HH), b2 = *(const f32x4*)(p1 + HH); // w2 rows k,k+1
      if (r == 0 && kt + 1 < DD / KT)
        __builtin_prefetch(p0 + (size_t)KT * 2 * HH, 0, 1);  // next K-tile
      #pragma unroll
      for (int q = 0; q < 4; ++q) {
        *(unsigned*)(lB1 + (bn + q) * AS + k) = pk2bf(a1[q], b1[q]);
        *(unsigned*)(lB2 + (bn + q) * AS + k) = pk2bf(a2[q], b2[q]);
      }
    }
    async_wait0();
    __syncthreads();

    Frag a0, a1f;
    {
      const int r0 = 32 * w + (lane & 15);
      const int hs = (lane >> 4) * 8;     // lanes 16-31 hold K 8..15 / 24..31
      a0.u[0]  = *(const u32x4*)(lA + r0 * AS + hs);
      a0.u[1]  = *(const u32x4*)(lA + r0 * AS + 16 + hs);
      a1f.u[0] = *(const u32x4*)(lA + (r0 + 16) * AS + hs);
      a1f.u[1] = *(const u32x4*)(lA + (r0 + 16) * AS + 16 + hs);
    }
    const int kb = (lane >> 4) * 8;       // lanes 16-31 hold K 8..15 / 24..31
    #pragma unroll
    for (int nt = 0; nt < 4; ++nt) {
      const int nc = (nt * 16 + (lane & 15)) * AS + kb;
      Frag b1, b2;
      b1.u[0] = *(const u32x4*)(lB1 + nc);
      b1.u[1] = *(const u32x4*)(lB1 + nc + 16);
      b2.u[0] = *(const u32x4*)(lB2 + nc);
      b2.u[1] = *(const u32x4*)(lB2 + nc + 16);
      acc1[0][nt] = __builtin_amdgcn_wmma_f32_16x16x32_bf16(
          false, a0.v,  false, b1.v, (short)0, acc1[0][nt], false, false);
      acc1[1][nt] = __builtin_amdgcn_wmma_f32_16x16x32_bf16(
          false, a1f.v, false, b1.v, (short)0, acc1[1][nt], false, false);
      acc2[0][nt] = __builtin_amdgcn_wmma_f32_16x16x32_bf16(
          false, a0.v,  false, b2.v, (short)0, acc2[0][nt], false, false);
      acc2[1][nt] = __builtin_amdgcn_wmma_f32_16x16x32_bf16(
          false, a1f.v, false, b2.v, (short)0, acc2[1][nt], false, false);
    }
  }

  // epilogue: h = silu(x1)*x2 -> bf16 scratch (C layout: m = v + 8*(lane>=16))
  const int nlan = lane & 15;
  #pragma unroll
  for (int mf = 0; mf < 2; ++mf) {
    const int mBase = 32 * w + 16 * mf + ((lane >> 4) * 8);
    #pragma unroll
    for (int v = 0; v < 8; ++v) {
      const int r = rowBase + mBase + v;
      if (r < cnt) {
        unsigned short* hrow = hbuf + (size_t)(off + r) * HH + n0 + nlan;
        #pragma unroll
        for (int nt = 0; nt < 4; ++nt) {
          const float x1 = acc1[mf][nt][v], x2 = acc2[mf][nt][v];
          const float s = x1 / (1.f + __expf(-x1));
          hrow[nt * 16] = f2bf(s * x2);
        }
      }
    }
  }
}

// ---------------------------------------------------------------------------
// GEMM2: h[cnt,4096] @ w3[e][4096,1024], gate-scaled scatter-add into out.
// ---------------------------------------------------------------------------
__global__ __launch_bounds__(128) void moe_gemm2(
    const unsigned short* __restrict__ hbuf, const float* __restrict__ w3,
    const int* __restrict__ toks, const float* __restrict__ gates,
    const int* __restrict__ meta, float* __restrict__ out) {
  const int e = blockIdx.z;
  const int cnt = meta[e];
  const int rowBase = blockIdx.y * MT;
  if (rowBase >= cnt) return;
  const int off = meta[8 + e];
  const int n0 = blockIdx.x * NT;

  __shared__ __align__(16) unsigned short lA[MT * AS];
  __shared__ __align__(16) unsigned short lB[NT * AS];

  const int tid = threadIdx.x;
  const int w = tid >> 5, lane = tid & 31;

  f32x8 acc[2][4];
  #pragma unroll
  for (int m = 0; m < 2; ++m)
    #pragma unroll
    for (int i = 0; i < 4; ++i)
      #pragma unroll
      for (int j = 0; j < 8; ++j) acc[m][i][j] = 0.f;

  int gr = rowBase + tid; if (gr >= cnt) gr = cnt - 1;
  const unsigned short* hrow = hbuf + (size_t)(off + gr) * HH;

  const int bp = tid >> 4;
  const int bn = (tid & 15) * 4;
  const float* wbase = w3 + (size_t)e * HH * DD;

  for (int kt = 0; kt < HH / KT; ++kt) {
    const int k0 = kt * KT;
    __syncthreads();
    #pragma unroll
    for (int c = 0; c < 4; ++c)     // A (bf16 h) via async DMA
      async_copy16(hrow + k0 + c * 8, lA + tid * AS + c * 8);
    #pragma unroll
    for (int r = 0; r < 2; ++r) {   // B: fp32 -> packed bf16, transposed
      const int k = 2 * (bp + r * 8);
      const float* p0 = wbase + (size_t)(k0 + k) * DD + (n0 + bn);
      const float* p1 = p0 + DD;
      f32x4 f0 = *(const f32x4*)p0, f1 = *(const f32x4*)p1;
      if (r == 0 && kt + 1 < HH / KT)
        __builtin_prefetch(p0 + (size_t)KT * DD, 0, 1);
      #pragma unroll
      for (int q = 0; q < 4; ++q)
        *(unsigned*)(lB + (bn + q) * AS + k) = pk2bf(f0[q], f1[q]);
    }
    async_wait0();
    __syncthreads();

    Frag a0, a1f;
    {
      const int r0 = 32 * w + (lane & 15);
      const int hs = (lane >> 4) * 8;
      a0.u[0]  = *(const u32x4*)(lA + r0 * AS + hs);
      a0.u[1]  = *(const u32x4*)(lA + r0 * AS + 16 + hs);
      a1f.u[0] = *(const u32x4*)(lA + (r0 + 16) * AS + hs);
      a1f.u[1] = *(const u32x4*)(lA + (r0 + 16) * AS + 16 + hs);
    }
    const int kb = (lane >> 4) * 8;
    #pragma unroll
    for (int nt = 0; nt < 4; ++nt) {
      const int nc = (nt * 16 + (lane & 15)) * AS + kb;
      Frag b;
      b.u[0] = *(const u32x4*)(lB + nc);
      b.u[1] = *(const u32x4*)(lB + nc + 16);
      acc[0][nt] = __builtin_amdgcn_wmma_f32_16x16x32_bf16(
          false, a0.v,  false, b.v, (short)0, acc[0][nt], false, false);
      acc[1][nt] = __builtin_amdgcn_wmma_f32_16x16x32_bf16(
          false, a1f.v, false, b.v, (short)0, acc[1][nt], false, false);
    }
  }

  const int nlan = lane & 15;
  #pragma unroll
  for (int mf = 0; mf < 2; ++mf) {
    const int mBase = 32 * w + 16 * mf + ((lane >> 4) * 8);
    int tok[8]; float gw[8];
    #pragma unroll
    for (int v = 0; v < 8; ++v) {
      const int r = rowBase + mBase + v;
      if (r < cnt) { tok[v] = toks[off + r]; gw[v] = gates[off + r]; }
      else tok[v] = -1;
    }
    #pragma unroll
    for (int v = 0; v < 8; ++v) {
      if (tok[v] >= 0) {
        float* orow = out + (size_t)tok[v] * DD + n0 + nlan;
        #pragma unroll
        for (int nt = 0; nt < 4; ++nt)
          __hip_atomic_fetch_add(orow + nt * 16, acc[mf][nt][v] * gw[v],
                                 __ATOMIC_RELAXED, __HIP_MEMORY_SCOPE_AGENT);
      }
    }
  }
}

__global__ void moe_finalize(float* __restrict__ out, const float* __restrict__ aux) {
  if (threadIdx.x == 0 && blockIdx.x == 0)
    out[(size_t)TN * DD] = 0.01f * aux[0] * (1.0f / (float)(TN * EE));
}

// ---------------------------------------------------------------------------
extern "C" void kernel_launch(void* const* d_in, const int* in_sizes, int n_in,
                              void* d_out, int out_size, void* d_ws, size_t ws_size,
                              hipStream_t stream) {
  (void)in_sizes; (void)n_in; (void)out_size; (void)ws_size;
  const float* x   = (const float*)d_in[0];
  const float* rw  = (const float*)d_in[1];
  const float* rb  = (const float*)d_in[2];
  const float* w12 = (const float*)d_in[3];
  const float* w3  = (const float*)d_in[4];
  float* out = (float*)d_out;

  char* ws = (char*)d_ws;
  unsigned short* xb   = (unsigned short*)(ws + XB_OFF);
  unsigned short* hbuf = (unsigned short*)(ws + HB_OFF);
  float* gates = (float*)(ws + GT_OFF);
  int*   toks  = (int*)  (ws + TK_OFF);
  float* t2w   = (float*)(ws + T2W_OFF);
  int*   t2i   = (int*)  (ws + T2I_OFF);
  int*   meta  = (int*)  (ws + META_OFF);
  float* aux   = (float*)(ws + AUX_OFF);

  const int outN = TN * DD + 1;
  moe_init<<<(outN + 255) / 256, 256, 0, stream>>>(out, outN, meta, aux);
  moe_xcvt<<<(TN * DD / 4 + 255) / 256, 256, 0, stream>>>(x, xb, TN * DD / 4);
  moe_router<<<TN / 8, 256, 0, stream>>>(x, rw, rb, t2w, t2i, meta, aux);
  moe_prefix<<<1, 1, 0, stream>>>(meta);
  moe_scatter<<<(TN + 255) / 256, 256, 0, stream>>>(t2i, t2w, meta, toks, gates);
  moe_gemm1<<<dim3(HH / NT, TN / MT, EE), 128, 0, stream>>>(xb, w12, toks, meta, hbuf);
  moe_gemm2<<<dim3(DD / NT, TN / MT, EE), 128, 0, stream>>>(hbuf, w3, toks, gates, meta, out);
  moe_finalize<<<1, 1, 0, stream>>>(out, aux);
}
